// PrototypeMatchingModel_16750372455063
// MI455X (gfx1250) — matrix operand order repaired and checked by reading, not compiled, not run on hardware
//
#include <hip/hip_runtime.h>
#include <hip/hip_bf16.h>

// ---------------------------------------------------------------------------
// Problem constants (from reference): x (B=32, C=256, HW=1024) f32,
// prototype_bank (P=512, C=256) f32.
// Outputs concatenated in d_out: rec (B,C,H,W) f32 [8388608],
// indices (B,HW) i32 [32768], counts (P,) i32 [512].
// ---------------------------------------------------------------------------
#define BDIM   32
#define CDIM   256
#define HWDIM  1024
#define PDIM   512
#define MDIM   (BDIM * HWDIM)          // 32768 rows
#define REC_ELEMS (BDIM * CDIM * HWDIM)
#define IDX_ELEMS (BDIM * HWDIM)
#define EPSN   1e-12f

typedef _Float16 v8h  __attribute__((ext_vector_type(8)));
typedef _Float16 v16h __attribute__((ext_vector_type(16)));
typedef float    v8f  __attribute__((ext_vector_type(8)));

__device__ __forceinline__ v16h cat16(v8h lo, v8h hi) {
    return __builtin_shufflevector(lo, hi, 0, 1, 2, 3, 4, 5, 6, 7,
                                           8, 9, 10, 11, 12, 13, 14, 15);
}

// ---------------------------------------------------------------------------
// Kernel 1: normalize prototype rows (P,C) f32 -> (P,C) f16 in ws.
// Row-major (P,C) is exactly WMMA-B column order (column n = prototype p).
// ---------------------------------------------------------------------------
__global__ __launch_bounds__(256) void proto_norm_kernel(
    const float* __restrict__ proto, _Float16* __restrict__ pB) {
    __shared__ float red[256];
    const int p = blockIdx.x;          // 512 blocks
    const int t = threadIdx.x;         // 256 threads: one per channel
    float v = proto[p * CDIM + t];
    red[t] = v * v;
    __syncthreads();
    for (int s = 128; s > 0; s >>= 1) {
        if (t < s) red[t] += red[t + s];
        __syncthreads();
    }
    float scale = 1.0f / fmaxf(sqrtf(red[0]), EPSN);
    pB[p * CDIM + t] = (_Float16)(v * scale);
}

// ---------------------------------------------------------------------------
// Kernel 2: normalize x over channels per (b,hw) -> A matrix (M,C) f16 in ws.
// x is (B, C, HW): channel stride = HW floats. Reads coalesced over hw.
// ---------------------------------------------------------------------------
__global__ __launch_bounds__(64) void x_norm_kernel(
    const float* __restrict__ x, _Float16* __restrict__ xA) {
    const int i = blockIdx.x * 64 + threadIdx.x;   // row id in [0, M)
    const int b = i >> 10;
    const int n = i & (HWDIM - 1);
    const long base = (long)b * CDIM * HWDIM + n;
    float s = 0.0f;
    for (int c = 0; c < CDIM; ++c) {
        float v = x[base + (long)c * HWDIM];
        s += v * v;
    }
    const float scale = 1.0f / fmaxf(sqrtf(s), EPSN);
    _Float16* dst = xA + (long)i * CDIM;           // contiguous per thread
    for (int c = 0; c < CDIM; ++c) {
        float v = x[base + (long)c * HWDIM];       // hot in L1/L2 (re-read)
        dst[c] = (_Float16)(v * scale);
    }
}

// ---------------------------------------------------------------------------
// Kernel 3: fused WMMA GEMM + argmax. Block = 128 threads (4 waves),
// wave owns a 16-row M tile (A resident in VGPRs), prototypes streamed
// through 32KB LDS chunks of 64 columns. Grid = M/64 = 512 blocks.
// ---------------------------------------------------------------------------
#define CHUNK_P  64
#define NTILES_PER_CHUNK (CHUNK_P / 16)
#define NCHUNKS  (PDIM / CHUNK_P)

__global__ __launch_bounds__(128) void match_kernel(
    const _Float16* __restrict__ xA, const _Float16* __restrict__ pB,
    int* __restrict__ idx_out, int* __restrict__ counts) {
    __shared__ __attribute__((aligned(32))) _Float16 sB[CHUNK_P * CDIM]; // 32KB

    const int tid  = threadIdx.x;
    const int wave = tid >> 5;
    const int lane = tid & 31;
    const int half = lane >> 4;        // 0: rows m..m+7 | 1: rows m+8..m+15
    const int col  = lane & 15;        // N column within 16-wide tile
    const int row_base = blockIdx.x * 64 + wave * 16;

    // --- Load resident A fragments: 8 K-steps x v16h (contiguous K layout:
    // lane<16 holds K 0..15, lane>=16 holds K 16..31) ---
    v16h afrag[8];
    {
        const _Float16* arow = xA + (long)(row_base + col) * CDIM;
        const int koff = half * 16;
        for (int ks = 0; ks < 8; ++ks) {
            v8h lo = *(const v8h*)(arow + ks * 32 + koff);
            v8h hi = *(const v8h*)(arow + ks * 32 + koff + 8);
            afrag[ks] = cat16(lo, hi);
        }
    }

    float best_v[8];
    int   best_i[8];
    #pragma unroll
    for (int r = 0; r < 8; ++r) { best_v[r] = -3.402823466e38f; best_i[r] = 0; }

    for (int chunk = 0; chunk < NCHUNKS; ++chunk) {
        // Cooperative global -> LDS staging of 64 prototypes (f16).
        __syncthreads();
        {
            const uint4* src = (const uint4*)(pB + chunk * CHUNK_P * CDIM);
            uint4* dst = (uint4*)sB;
            #pragma unroll
            for (int j = 0; j < (CHUNK_P * CDIM * 2 / 16) / 128; ++j)
                dst[tid + j * 128] = src[tid + j * 128];
        }
        if (chunk + 1 < NCHUNKS)       // speculative prefetch of next chunk
            __builtin_prefetch(pB + (chunk + 1) * CHUNK_P * CDIM + tid * 128, 0, 0);
        __syncthreads();

        for (int nt = 0; nt < NTILES_PER_CHUNK; ++nt) {
            v8f acc = {};
            const int p_local = nt * 16 + col;
            // B layout: lanes 0-15 hold K 0..15 of column `col`,
            // lanes 16-31 hold K 16..31 (contiguous 32B in LDS per lane).
            const _Float16* bcol = sB + p_local * CDIM + half * 16;
            #pragma unroll
            for (int ks = 0; ks < 8; ++ks) {
                v16h bfrag = *(const v16h*)(bcol + ks * 32);
                acc = __builtin_amdgcn_wmma_f32_16x16x32_f16(
                    false, afrag[ks], false, bfrag, (short)0, acc, false, false);
            }
            const int p_global = chunk * CHUNK_P + nt * 16 + col;
            #pragma unroll
            for (int r = 0; r < 8; ++r) {
                if (acc[r] > best_v[r]) { best_v[r] = acc[r]; best_i[r] = p_global; }
            }
        }
    }

    // Argmax across the 16 N-columns held in each 16-lane half.
    // xor masks 1,2,4,8 stay within the half. First-occurrence tie-break.
    #pragma unroll
    for (int r = 0; r < 8; ++r) {
        float bv = best_v[r];
        int   bi = best_i[r];
        #pragma unroll
        for (int mask = 8; mask >= 1; mask >>= 1) {
            float ov = __shfl_xor(bv, mask, 32);
            int   oi = __shfl_xor(bi, mask, 32);
            if (ov > bv || (ov == bv && oi < bi)) { bv = ov; bi = oi; }
        }
        if (col == 0) {                 // lane 0 -> rows 0..7, lane 16 -> 8..15
            const int m = row_base + half * 8 + r;
            idx_out[m] = bi;
            atomicAdd(&counts[bi], 1);
        }
    }
}

// ---------------------------------------------------------------------------
// Kernel 4: rec[b,c,hw] = proto[idx[b,hw], c]  (coalesced stores over hw).
// ---------------------------------------------------------------------------
__global__ __launch_bounds__(256) void writeout_kernel(
    const float* __restrict__ proto, const int* __restrict__ idx,
    float* __restrict__ rec) {
    const long tid = (long)blockIdx.x * 256 + threadIdx.x;  // < REC_ELEMS
    const int  b   = (int)(tid >> 18);          // C*HW = 262144 = 2^18
    const int  rem = (int)(tid & 262143);
    const int  c   = rem >> 10;
    const int  hw  = rem & 1023;
    const int  p   = idx[b * HWDIM + hw];
    rec[tid] = proto[p * CDIM + c];
}

// ---------------------------------------------------------------------------
extern "C" void kernel_launch(void* const* d_in, const int* in_sizes, int n_in,
                              void* d_out, int out_size, void* d_ws, size_t ws_size,
                              hipStream_t stream) {
    const float* x     = (const float*)d_in[0];   // (32,256,32,32)
    const float* proto = (const float*)d_in[1];   // (512,256)

    float* rec    = (float*)d_out;
    int*   idx    = (int*)(rec + REC_ELEMS);
    int*   counts = (int*)(rec + REC_ELEMS + IDX_ELEMS);

    // Workspace layout: A matrix f16 (M x C) then normalized protos f16 (P x C).
    _Float16* xA = (_Float16*)d_ws;                       // 16 MB
    _Float16* pB = (_Float16*)((char*)d_ws + (size_t)MDIM * CDIM * 2);

    hipMemsetAsync(counts, 0, PDIM * sizeof(int), stream);

    proto_norm_kernel<<<PDIM, 256, 0, stream>>>(proto, pB);
    x_norm_kernel<<<MDIM / 64, 64, 0, stream>>>(x, xA);
    match_kernel<<<MDIM / 64, 128, 0, stream>>>(xA, pB, idx, counts);
    writeout_kernel<<<REC_ELEMS / 256, 256, 0, stream>>>(proto, idx, rec);
}
